// MultiHeadAttention_62440234549650
// MI455X (gfx1250) — matrix operand order repaired and checked by reading, not compiled, not run on hardware
//
#include <hip/hip_runtime.h>

typedef __attribute__((ext_vector_type(16))) __bf16 v16bf;
typedef __attribute__((ext_vector_type(8)))  float  v8f;
typedef int v4i_ __attribute__((vector_size(16)));

struct alignas(16) U4 { unsigned x, y, z, w; };
struct alignas(16) F4 { float x, y, z, w; };
union AFrag { v16bf v; U4 u[2]; };

constexpr int HID    = 1024;
constexpr int SEQ    = 2048;
constexpr int NHEAD  = 16;
constexpr int HDIM   = 64;
constexpr int NBATCH = 2;
constexpr int MTOT   = NBATCH * SEQ;   // 4096
constexpr float LOG2E = 1.4426950408889634f;

#if defined(__AMDGCN__) && __has_builtin(__builtin_amdgcn_global_load_async_to_lds_b128)
#define HAS_ASYNC_LDS 1
#else
#define HAS_ASYNC_LDS 0
#endif

// ---- bf16 conversion: prefer v_cvt_pk_bf16_f32 when the builtin exists ----
#if defined(__AMDGCN__) && __has_builtin(__builtin_amdgcn_cvt_pk_bf16_f32)
typedef __attribute__((ext_vector_type(2))) __bf16 v2bf;
__device__ inline unsigned pack2(float lo, float hi) {
  union { v2bf v; unsigned u; } c;
  c.v = __builtin_amdgcn_cvt_pk_bf16_f32(lo, hi);
  return c.u;
}
__device__ inline unsigned short f2bf(float f) {
  return (unsigned short)(pack2(f, 0.f) & 0xffffu);
}
#else
__device__ inline unsigned short f2bf(float f) {
  union { float f; unsigned u; } c; c.f = f;
  unsigned u = c.u;
  u += 0x7fffu + ((u >> 16) & 1u);     // round-to-nearest-even
  return (unsigned short)(u >> 16);
}
__device__ inline unsigned pack2(float lo, float hi) {
  return (unsigned)f2bf(lo) | ((unsigned)f2bf(hi) << 16);
}
#endif

__device__ inline float redmax16(float v) {
  v = fmaxf(v, __shfl_xor(v, 1));
  v = fmaxf(v, __shfl_xor(v, 2));
  v = fmaxf(v, __shfl_xor(v, 4));
  v = fmaxf(v, __shfl_xor(v, 8));
  return v;
}
__device__ inline float redsum16(float v) {
  v += __shfl_xor(v, 1);
  v += __shfl_xor(v, 2);
  v += __shfl_xor(v, 4);
  v += __shfl_xor(v, 8);
  return v;
}

#if HAS_ASYNC_LDS
__device__ inline void async_ld_b128(const void* g, void* l) {
  __builtin_amdgcn_global_load_async_to_lds_b128(
      (__attribute__((address_space(1))) v4i_*)(g),
      (__attribute__((address_space(3))) v4i_*)(l), 0, 0);
}
template <int N> __device__ inline void wait_async() {
#if __has_builtin(__builtin_amdgcn_s_wait_asynccnt)
  __builtin_amdgcn_s_wait_asynccnt(N);
#else
  if constexpr (N == 0) asm volatile("s_wait_asynccnt 0x0" ::: "memory");
  else                  asm volatile("s_wait_asynccnt 0x4" ::: "memory");
#endif
}
#endif

// MODE: 0=Q (bf16 [b,h,s,d], *0.125), 1=K (bf16 [b,h,s,d]),
//       2=V transposed (bf16 [b,h,d,s]), 3=final output (fp32 [b,s,hid] + bias)
// WG tile: 128 (M) x 128 (N); wave = 32 rows x 64 cols -> 8 WMMAs / k-step.
template <int MODE>
__global__ __launch_bounds__(256)
void proj_kernel(const void* __restrict__ Ap, const float* __restrict__ W,
                 const float* __restrict__ bias, void* __restrict__ Outp) {
  __shared__ alignas(16) unsigned short Asm_[128 * 32];  // 8 KB, [row][k]
  __shared__ alignas(16) unsigned short Bsm_[128 * 32];  // 8 KB, [n][k]

  const int tid  = threadIdx.x;
  const int wave = tid >> 5, lane = tid & 31, hf = lane >> 4, ln = lane & 15;
  const int rw = wave & 3, cw = wave >> 2;   // 4 waves in M, 2 in N
  const int gm0 = blockIdx.x * 128;
  const int gn0 = blockIdx.y * 128;

  const v8f vz = {0.f,0.f,0.f,0.f,0.f,0.f,0.f,0.f};
  v8f acc[2][4];
#pragma unroll
  for (int a = 0; a < 2; ++a)
#pragma unroll
    for (int t = 0; t < 4; ++t) acc[a][t] = vz;

  // staging: both tiles are 128 rows x 32 k -> 16 elems / thread
  const int rr = tid >> 1;
  const int cc = (tid & 1) * 16;

  U4 sa0, sa1, sb0, sb1;   // register-staged bf16 data (software pipeline)
  auto loadA = [&](int kb) {
    if (MODE == 3) {       // A is bf16 ctx
      const unsigned short* A = (const unsigned short*)Ap + (size_t)(gm0 + rr) * HID + kb + cc;
      sa0 = *(const U4*)A; sa1 = *(const U4*)(A + 8);
    } else {               // A is fp32 x: convert in registers
      const float* A = (const float*)Ap + (size_t)(gm0 + rr) * HID + kb + cc;
      F4 a0 = *(const F4*)A;        F4 a1 = *(const F4*)(A + 4);
      F4 a2 = *(const F4*)(A + 8);  F4 a3 = *(const F4*)(A + 12);
      sa0 = {pack2(a0.x,a0.y), pack2(a0.z,a0.w), pack2(a1.x,a1.y), pack2(a1.z,a1.w)};
      sa1 = {pack2(a2.x,a2.y), pack2(a2.z,a2.w), pack2(a3.x,a3.y), pack2(a3.z,a3.w)};
    }
  };
  auto loadB = [&](int kb) {
    const float* Wp = W + (size_t)(gn0 + rr) * HID + kb + cc;
    F4 b0 = *(const F4*)Wp;        F4 b1 = *(const F4*)(Wp + 4);
    F4 b2 = *(const F4*)(Wp + 8);  F4 b3 = *(const F4*)(Wp + 12);
    sb0 = {pack2(b0.x,b0.y), pack2(b0.z,b0.w), pack2(b1.x,b1.y), pack2(b1.z,b1.w)};
    sb1 = {pack2(b2.x,b2.y), pack2(b2.z,b2.w), pack2(b3.x,b3.y), pack2(b3.z,b3.w)};
  };

  loadA(0); loadB(0);
  for (int kb = 0; kb < HID; kb += 32) {
    __syncthreads();                 // previous iteration's frag reads done
    *(U4*)&Asm_[rr * 32 + cc]     = sa0;
    *(U4*)&Asm_[rr * 32 + cc + 8] = sa1;
    *(U4*)&Bsm_[rr * 32 + cc]     = sb0;
    *(U4*)&Bsm_[rr * 32 + cc + 8] = sb1;
    __syncthreads();
    if (kb + 32 < HID) { loadA(kb + 32); loadB(kb + 32); }  // overlap with WMMA

    AFrag af[2], bfr[4];
#pragma unroll
    for (int a = 0; a < 2; ++a) {
      const int row = 32 * rw + 16 * a + ln;
      af[a].u[0] = *(const U4*)&Asm_[row * 32 + 8 * hf];
      af[a].u[1] = *(const U4*)&Asm_[row * 32 + 16 + 8 * hf];
    }
#pragma unroll
    for (int t = 0; t < 4; ++t) {
      const int n = 64 * cw + 16 * t + ln;
      bfr[t].u[0] = *(const U4*)&Bsm_[n * 32 + 8 * hf];
      bfr[t].u[1] = *(const U4*)&Bsm_[n * 32 + 16 + 8 * hf];
    }
#pragma unroll
    for (int a = 0; a < 2; ++a)
#pragma unroll
      for (int t = 0; t < 4; ++t)
        acc[a][t] = __builtin_amdgcn_wmma_f32_16x16x32_bf16(false, af[a].v, false, bfr[t].v,
                                                            (short)0, acc[a][t], false, false);
  }

  float bvt[4];
#pragma unroll
  for (int t = 0; t < 4; ++t) bvt[t] = bias[gn0 + 64 * cw + 16 * t + ln];

  if (MODE == 0 || MODE == 1) {
    unsigned short* O = (unsigned short*)Outp;
    const float scl = (MODE == 0) ? 0.125f : 1.0f;   // fold 1/sqrt(64) into Q
#pragma unroll
    for (int a = 0; a < 2; ++a)
#pragma unroll
      for (int t = 0; t < 4; ++t) {
        const int col = gn0 + 64 * cw + 16 * t + ln;
        const int h = col >> 6, d = col & 63;
#pragma unroll
        for (int i = 0; i < 8; ++i) {
          const int rowg = gm0 + 32 * rw + 16 * a + 8 * hf + i;
          const int b = rowg >> 11, s = rowg & (SEQ - 1);
          O[((size_t)(b * NHEAD + h) * SEQ + s) * HDIM + d] = f2bf((acc[a][t][i] + bvt[t]) * scl);
        }
      }
  } else if (MODE == 2) {  // V transposed: consecutive s per lane -> one b128 store
    unsigned short* O = (unsigned short*)Outp;
#pragma unroll
    for (int a = 0; a < 2; ++a) {
      const int rb = gm0 + 32 * rw + 16 * a + 8 * hf;
      const int b = rb >> 11, s0 = rb & (SEQ - 1);
#pragma unroll
      for (int t = 0; t < 4; ++t) {
        const int col = gn0 + 64 * cw + 16 * t + ln;
        const int h = col >> 6, d = col & 63;
        U4 p;
        p.x = pack2(acc[a][t][0] + bvt[t], acc[a][t][1] + bvt[t]);
        p.y = pack2(acc[a][t][2] + bvt[t], acc[a][t][3] + bvt[t]);
        p.z = pack2(acc[a][t][4] + bvt[t], acc[a][t][5] + bvt[t]);
        p.w = pack2(acc[a][t][6] + bvt[t], acc[a][t][7] + bvt[t]);
        *(U4*)&O[((size_t)(b * NHEAD + h) * HDIM + d) * SEQ + s0] = p;
      }
    }
  } else {
    float* O = (float*)Outp;
#pragma unroll
    for (int a = 0; a < 2; ++a)
#pragma unroll
      for (int t = 0; t < 4; ++t) {
        const int col = gn0 + 64 * cw + 16 * t + ln;
#pragma unroll
        for (int i = 0; i < 8; ++i) {
          const int rowg = gm0 + 32 * rw + 16 * a + 8 * hf + i;
          O[(size_t)rowg * HID + col] = acc[a][t][i] + bvt[t];
        }
      }
  }
}

// Flash attention: one WG = 128 queries of one (b,h); 8 waves x 16 queries.
// K/V tiles double-buffered in LDS via ASYNCcnt-tracked async copies.
__global__ __launch_bounds__(256)
void attn_kernel(const unsigned short* __restrict__ Q, const unsigned short* __restrict__ K,
                 const unsigned short* __restrict__ Vt, unsigned short* __restrict__ Ctx) {
  __shared__ alignas(16) unsigned short Ksm[2][64 * 64];   // [buf][key][d]   16 KB
  __shared__ alignas(16) unsigned short Vsm[2][64 * 64];   // [buf][d][key]   16 KB
  __shared__ alignas(16) unsigned short Psm[8 * 16 * 64];  // per-wave P      16 KB

  const int tid  = threadIdx.x;
  const int wave = tid >> 5, lane = tid & 31, hf = lane >> 4, ln = lane & 15;
  const int bh = blockIdx.x / (SEQ / 128);
  const int qb = blockIdx.x % (SEQ / 128);
  const int b = bh >> 4, h = bh & 15;

  AFrag qf[2];   // 16 x 64 Q tile (scale pre-folded), two k-steps of 32
  {
    const unsigned short* qp = Q + ((size_t)bh * SEQ + qb * 128 + 16 * wave + ln) * HDIM;
    qf[0].u[0] = *(const U4*)(qp + 8 * hf);
    qf[0].u[1] = *(const U4*)(qp + 16 + 8 * hf);
    qf[1].u[0] = *(const U4*)(qp + 32 + 8 * hf);
    qf[1].u[1] = *(const U4*)(qp + 48 + 8 * hf);
  }

  const v8f vz = {0.f,0.f,0.f,0.f,0.f,0.f,0.f,0.f};
  float mrow[8], lrow[8];
  v8f accd[4];
#pragma unroll
  for (int i = 0; i < 8; ++i) { mrow[i] = -3.0e38f; lrow[i] = 0.f; }
#pragma unroll
  for (int t = 0; t < 4; ++t) accd[t] = vz;

  const int sr = tid >> 2;          // staging: 64 rows x 64 cols, 32 B/thread each tile
  const int sc = (tid & 3) * 16;

  auto issue_tile = [&](int kb, int buf) {   // 4 x b128 per thread
    const unsigned short* kp = K + ((size_t)bh * SEQ + kb + sr) * HDIM + sc;
    const unsigned short* vp = Vt + ((size_t)bh * HDIM + sr) * SEQ + kb + sc;
#if HAS_ASYNC_LDS
    async_ld_b128(kp,     &Ksm[buf][sr * 64 + sc]);
    async_ld_b128(kp + 8, &Ksm[buf][sr * 64 + sc + 8]);
    async_ld_b128(vp,     &Vsm[buf][sr * 64 + sc]);
    async_ld_b128(vp + 8, &Vsm[buf][sr * 64 + sc + 8]);
#else
    *(U4*)&Ksm[buf][sr * 64 + sc]     = *(const U4*)kp;
    *(U4*)&Ksm[buf][sr * 64 + sc + 8] = *(const U4*)(kp + 8);
    *(U4*)&Vsm[buf][sr * 64 + sc]     = *(const U4*)vp;
    *(U4*)&Vsm[buf][sr * 64 + sc + 8] = *(const U4*)(vp + 8);
#endif
  };

  constexpr int NT = SEQ / 64;   // 32 key tiles
#if HAS_ASYNC_LDS
  issue_tile(0, 0);
#endif

  for (int it = 0; it < NT; ++it) {
    const int cur = it & 1;
#if HAS_ASYNC_LDS
    __syncthreads();                              // all reads of buf cur^1 finished
    if (it + 1 < NT) { issue_tile(64 * (it + 1), cur ^ 1); wait_async<4>(); }
    else             { wait_async<0>(); }
    __syncthreads();                              // tile `it` visible to all waves
#else
    __syncthreads();
    issue_tile(64 * it, cur);
    __syncthreads();
#endif
    const unsigned short* Kt = &Ksm[cur][0];
    const unsigned short* Vm = &Vsm[cur][0];

    // scores: S = Q (16x64) * K^T, 8 WMMAs
    v8f scv[4];
#pragma unroll
    for (int t = 0; t < 4; ++t) scv[t] = vz;
#pragma unroll
    for (int t = 0; t < 4; ++t) {
      const int key = 16 * t + ln;
#pragma unroll
      for (int kk = 0; kk < 2; ++kk) {
        AFrag bf_;
        bf_.u[0] = *(const U4*)&Kt[key * 64 + 32 * kk + 8 * hf];
        bf_.u[1] = *(const U4*)&Kt[key * 64 + 32 * kk + 16 + 8 * hf];
        scv[t] = __builtin_amdgcn_wmma_f32_16x16x32_bf16(false, qf[kk].v, false, bf_.v,
                                                         (short)0, scv[t], false, false);
      }
    }

    // online softmax (row = i + 8*hf, spread over 16 lanes of the half-wave)
#pragma unroll
    for (int i = 0; i < 8; ++i) {
      float tm = fmaxf(fmaxf(scv[0][i], scv[1][i]), fmaxf(scv[2][i], scv[3][i]));
      tm = redmax16(tm);
      const float mnew  = fmaxf(mrow[i], tm);
      const float alpha = exp2f((mrow[i] - mnew) * LOG2E);
      float rs = 0.f;
#pragma unroll
      for (int t = 0; t < 4; ++t) {
        const float p = exp2f((scv[t][i] - mnew) * LOG2E);
        scv[t][i] = p;
        rs += p;
      }
      rs = redsum16(rs);
      lrow[i] = lrow[i] * alpha + rs;
      mrow[i] = mnew;
#pragma unroll
      for (int t = 0; t < 4; ++t) accd[t][i] *= alpha;
    }

    // C-layout -> A-fragment layout via per-wave LDS round trip (in-order DS)
    unsigned short* P = &Psm[wave * (16 * 64)];
#pragma unroll
    for (int t = 0; t < 4; ++t)
#pragma unroll
      for (int i = 0; i < 8; ++i)
        P[(8 * hf + i) * 64 + 16 * t + ln] = f2bf(scv[t][i]);

    AFrag pf[2];
#pragma unroll
    for (int kk = 0; kk < 2; ++kk) {
      pf[kk].u[0] = *(const U4*)&P[ln * 64 + 32 * kk + 8 * hf];
      pf[kk].u[1] = *(const U4*)&P[ln * 64 + 32 * kk + 16 + 8 * hf];
    }

    // ctx += P (16x64) * V, 8 WMMAs; B-frag contiguous thanks to transposed V
#pragma unroll
    for (int t = 0; t < 4; ++t) {
      const int dim = 16 * t + ln;
#pragma unroll
      for (int kk = 0; kk < 2; ++kk) {
        AFrag bf_;
        bf_.u[0] = *(const U4*)&Vm[dim * 64 + 32 * kk + 8 * hf];
        bf_.u[1] = *(const U4*)&Vm[dim * 64 + 32 * kk + 16 + 8 * hf];
        accd[t] = __builtin_amdgcn_wmma_f32_16x16x32_bf16(false, pf[kk].v, false, bf_.v,
                                                          (short)0, accd[t], false, false);
      }
    }
  }

  float rinv[8];
#pragma unroll
  for (int i = 0; i < 8; ++i) rinv[i] = 1.0f / lrow[i];
#pragma unroll
  for (int t = 0; t < 4; ++t)
#pragma unroll
    for (int i = 0; i < 8; ++i) {
      const int qg = qb * 128 + 16 * wave + 8 * hf + i;
      Ctx[((size_t)b * SEQ + qg) * HID + h * 64 + 16 * t + ln] = f2bf(accd[t][i] * rinv[i]);
    }
}

extern "C" void kernel_launch(void* const* d_in, const int* in_sizes, int n_in,
                              void* d_out, int out_size, void* d_ws, size_t ws_size,
                              hipStream_t stream) {
  (void)in_sizes; (void)n_in; (void)out_size; (void)ws_size;
  const float* x  = (const float*)d_in[0];
  const float* wq = (const float*)d_in[1];
  const float* bq = (const float*)d_in[2];
  const float* wk = (const float*)d_in[3];
  const float* bk = (const float*)d_in[4];
  const float* wv = (const float*)d_in[5];
  const float* bv = (const float*)d_in[6];
  const float* wo = (const float*)d_in[7];
  const float* bo = (const float*)d_in[8];
  float* out = (float*)d_out;

  const size_t QSZ = (size_t)MTOT * HID;   // 4 Mi elements each
  unsigned short* Qb = (unsigned short*)d_ws;
  unsigned short* Kb = Qb + QSZ;
  unsigned short* Vb = Kb + QSZ;
  unsigned short* Cb = Vb + QSZ;           // 32 MB total workspace

  dim3 blk(256);
  dim3 gproj(MTOT / 128, HID / 128);       // 32 x 8
  proj_kernel<0><<<gproj, blk, 0, stream>>>((const void*)x, wq, bq, (void*)Qb);
  proj_kernel<1><<<gproj, blk, 0, stream>>>((const void*)x, wk, bk, (void*)Kb);
  proj_kernel<2><<<gproj, blk, 0, stream>>>((const void*)x, wv, bv, (void*)Vb);
  attn_kernel<<<dim3(NBATCH * NHEAD * (SEQ / 128)), blk, 0, stream>>>(Qb, Kb, Vb, Cb);
  proj_kernel<3><<<gproj, blk, 0, stream>>>((const void*)Cb, wo, bo, (void*)out);
}
